// UnidirectionalAttention_28183575396360
// MI455X (gfx1250) — matrix operand order repaired and checked by reading, not compiled, not run on hardware
//
#include <hip/hip_runtime.h>

// ---------------- problem constants ----------------
#define LDIM   200
#define SDIM   81          // 9*9
#define CENTER 40
#define DKDIM  128
#define HEADS  4
#define DHEAD  32
#define MASK_VAL  (-1000000.0f)
#define SCALE     0.08838834764831845f   // 1/sqrt(128)
#define XELEMS    (LDIM * SDIM)          // 16200

// ---------------- tiling ----------------
#define KC_COUNT 7         // K padded 200 -> 224, 32 per WMMA
#define NT_COUNT 6         // N padded 81 -> 96, 16 per tile
#define QNT      2         // N-tile containing CENTER (cols 32..47)
#define QLH      8         // CENTER - 32
#define WSTRIDE  224       // bf16 elems per weight row in LDS (K-padded)
#define XSTRIDE  224       // bf16 elems per X^T row (one S column) in LDS
#define NPAD     96
#define RSTRIDE  84        // f32 result row stride (cols 0..80 valid)

#define NBLOCKS  2048      // persistent blocks; each handles B/NBLOCKS batches

typedef __bf16 bf16_t;
typedef __bf16 v16bf __attribute__((ext_vector_type(16)));
typedef __bf16 v8bf  __attribute__((ext_vector_type(8)));
typedef float  v8f   __attribute__((ext_vector_type(8)));

// ---------------- LDS layout (bytes) ----------------
#define OFF_WK   0                                   // bf16 [128][224]
#define OFF_WV   (128 * WSTRIDE * 2)                 // 57344
#define OFF_WQ   (2 * 128 * WSTRIDE * 2)             // 114688
#define OFF_XT   (3 * 128 * WSTRIDE * 2)             // 172032, bf16 [96][224]
#define OFF_KT   (OFF_XT + NPAD * XSTRIDE * 2)       // 215040, f32 [128][84]
#define OFF_VT   (OFF_KT + DKDIM * RSTRIDE * 4)      // 258048, f32 [128][84]
#define OFF_Q    (OFF_VT + DKDIM * RSTRIDE * 4)      // 301056, f32[128]
#define OFF_SC   (OFF_Q  + 512)                      // f32[4*96]
#define OFF_BK   (OFF_SC + 4 * 96 * 4)               // f32[128]
#define OFF_BV   (OFF_BK + 512)                      // f32[128]
#define OFF_BQ   (OFF_BV + 512)                      // f32[128]
#define OFF_ST   (OFF_BQ + 512)                      // f32[8]
#define OFF_RED  (OFF_ST + 32)                       // f32[128]
#define OFF_GATE (OFF_RED + 512)                     // f32[1]
#define SMEM_BYTES (OFF_GATE + 16)                   // ~305 KB (< 320 KB/WGP)

// element-unit offsets between co-based weight regions
#define WV_EOFF  (OFF_WV / 2)                        // 28672 bf16 elems
#define WQ_EOFF  (OFF_WQ / 2)                        // 57344 bf16 elems

__device__ __forceinline__ bf16_t f2bf(float f) {
    union { float f; unsigned u; } c; c.f = f;
    unsigned r = (c.u + 0x7FFFu + ((c.u >> 16) & 1u)) >> 16;
    union { unsigned short s; bf16_t b; } o; o.s = (unsigned short)r;
    return o.b;
}

__global__ __launch_bounds__(256, 1)
void surround_attn_kernel(const float* __restrict__ x,
                          const float* __restrict__ Wk, const float* __restrict__ bk,
                          const float* __restrict__ Wv, const float* __restrict__ bv,
                          const float* __restrict__ Wq, const float* __restrict__ bq,
                          const float* __restrict__ Wg, const float* __restrict__ bg,
                          float* __restrict__ out, int nbatch)
{
    __shared__ char smem[SMEM_BYTES] __attribute__((aligned(16)));

    const int tid  = threadIdx.x;
    const int lane = tid & 31;
    const int wave = tid >> 5;
    const int lh   = lane & 15;
    const int sel  = lane >> 4;

    // ---- one-time: zero weight + X^T regions (establishes all K/N zero-pads) ----
    {
        uint4  z; z.x = z.y = z.z = z.w = 0u;
        uint4* p = (uint4*)smem;
        const int n16 = OFF_KT / 16;     // covers Wk,Wv,Wq,XT
        for (int i = tid; i < n16; i += 256) p[i] = z;
    }
    __syncthreads();

    // ---- one-time: stage Wk/Wv/Wq as bf16, biases as f32 ----
    {
        bf16_t* wk = (bf16_t*)(smem + OFF_WK);
        bf16_t* wv = (bf16_t*)(smem + OFF_WV);
        bf16_t* wq = (bf16_t*)(smem + OFF_WQ);
        for (int row = wave; row < DKDIM; row += 8)
            for (int l = lane; l < LDIM; l += 32) {
                wk[row * WSTRIDE + l] = f2bf(Wk[row * LDIM + l]);
                wv[row * WSTRIDE + l] = f2bf(Wv[row * LDIM + l]);
                wq[row * WSTRIDE + l] = f2bf(Wq[row * LDIM + l]);
            }
        if (tid < DKDIM) {
            ((float*)(smem + OFF_BK))[tid] = bk[tid];
            ((float*)(smem + OFF_BV))[tid] = bv[tid];
            ((float*)(smem + OFF_BQ))[tid] = bq[tid];
        }
    }
    __syncthreads();

    // ---- hoisted LDS base pointers: every ds_load below is base + imm offset ----
    const bf16_t* paK = (const bf16_t*)(smem + OFF_WK) + (wave * 16 + lh) * WSTRIDE + sel * 8;
    const bf16_t* paQ = paK + WQ_EOFF;   // WQ region is beyond 16-bit imm range
    const bf16_t* pB  = (const bf16_t*)(smem + OFF_XT) + lh * XSTRIDE + sel * 8;

    bf16_t* xt  = (bf16_t*)(smem + OFF_XT);
    float*  kt  = (float*)(smem + OFF_KT);
    float*  vt  = (float*)(smem + OFF_VT);
    float*  qv  = (float*)(smem + OFF_Q);
    float*  sc  = (float*)(smem + OFF_SC);
    float*  bkl = (float*)(smem + OFF_BK);
    float*  bvl = (float*)(smem + OFF_BV);
    float*  bql = (float*)(smem + OFF_BQ);
    float*  st  = (float*)(smem + OFF_ST);
    float*  red = (float*)(smem + OFF_RED);

    // ================= persistent batch loop =================
    for (int b = blockIdx.x; b < nbatch; b += (int)gridDim.x) {

        // ---- stage x[b] transposed -> bf16 X^T (vector loads) ----
        const float4* xb4 = (const float4*)(x + (size_t)b * XELEMS);
        for (int j = tid; j < XELEMS / 4; j += 256) {     // 4050 float4's
            float4 v = xb4[j];
            const float vf[4] = { v.x, v.y, v.z, v.w };
            const int i0 = 4 * j;
#pragma unroll
            for (int e = 0; e < 4; ++e) {
                const int i = i0 + e;
                const int l = i / SDIM;
                const int s = i - l * SDIM;
                xt[s * XSTRIDE + l] = f2bf(vf[e]);        // X^T[s][l]
            }
        }
        // ---- prefetch next batch's x into cache (global_prefetch_b8) ----
        {
            const int bn = b + (int)gridDim.x;
            if (bn < nbatch) {
                const char* nx = (const char*)(x + (size_t)bn * XELEMS);
                for (int j = tid; j < 507; j += 256)      // 64800B / 128B lines
                    __builtin_prefetch(nx + (size_t)j * 128, 0, 3);
            }
        }
        __syncthreads();

        // ---- GEMMs: keysT = Wk@X, valsT = Wv@X, plus query tile Wq@X[:,32..47] ----
        v8f acck[NT_COUNT], accv[NT_COUNT], accq;
#pragma unroll
        for (int nt = 0; nt < NT_COUNT; ++nt) {
            acck[nt] = (v8f){0.f,0.f,0.f,0.f,0.f,0.f,0.f,0.f};
            accv[nt] = (v8f){0.f,0.f,0.f,0.f,0.f,0.f,0.f,0.f};
        }
        accq = (v8f){0.f,0.f,0.f,0.f,0.f,0.f,0.f,0.f};
#pragma unroll
        for (int kc = 0; kc < KC_COUNT; ++kc) {
            const int ke = kc * 32;   // element offset within a row
            v8bf ak0 = *(const v8bf*)(paK + ke);
            v8bf ak1 = *(const v8bf*)(paK + ke + 16);
            v16bf Ak = __builtin_shufflevector(ak0, ak1, 0,1,2,3,4,5,6,7,8,9,10,11,12,13,14,15);
            v8bf av0 = *(const v8bf*)(paK + WV_EOFF + ke);
            v8bf av1 = *(const v8bf*)(paK + WV_EOFF + ke + 16);
            v16bf Av = __builtin_shufflevector(av0, av1, 0,1,2,3,4,5,6,7,8,9,10,11,12,13,14,15);
            v8bf aq0 = *(const v8bf*)(paQ + ke);
            v8bf aq1 = *(const v8bf*)(paQ + ke + 16);
            v16bf Aq = __builtin_shufflevector(aq0, aq1, 0,1,2,3,4,5,6,7,8,9,10,11,12,13,14,15);
#pragma unroll
            for (int nt = 0; nt < NT_COUNT; ++nt) {
                v8bf b0 = *(const v8bf*)(pB + nt * (16 * XSTRIDE) + ke);
                v8bf b1 = *(const v8bf*)(pB + nt * (16 * XSTRIDE) + ke + 16);
                v16bf Bf = __builtin_shufflevector(b0, b1, 0,1,2,3,4,5,6,7,8,9,10,11,12,13,14,15);
                acck[nt] = __builtin_amdgcn_wmma_f32_16x16x32_bf16(
                    false, Ak, false, Bf, (short)0, acck[nt], false, false);
                accv[nt] = __builtin_amdgcn_wmma_f32_16x16x32_bf16(
                    false, Av, false, Bf, (short)0, accv[nt], false, false);
                if (nt == QNT)
                    accq = __builtin_amdgcn_wmma_f32_16x16x32_bf16(
                        false, Aq, false, Bf, (short)0, accq, false, false);
            }
        }
        __syncthreads();

        // ---- spill: D layout VGPR r -> M = r + 8*sel, N = lh ----
#pragma unroll
        for (int nt = 0; nt < NT_COUNT; ++nt) {
            const int s = nt * 16 + lh;
            if (s < SDIM) {
#pragma unroll
                for (int r = 0; r < 8; ++r) {
                    const int d = wave * 16 + r + sel * 8;
                    kt[d * RSTRIDE + s] = acck[nt][r];
                    vt[d * RSTRIDE + s] = accv[nt][r];
                }
            }
        }
        if (lh == QLH) {   // lanes holding column CENTER of the query tile
#pragma unroll
            for (int r = 0; r < 8; ++r) {
                const int d = wave * 16 + r + sel * 8;
                qv[d] = accq[r] + bql[d];
            }
        }
        __syncthreads();

        // ---- scores[h][s] = (q_h . (k_s + bk)) * SCALE ; mask center ----
        for (int idx = tid; idx < HEADS * NPAD; idx += 256) {
            const int h = idx / NPAD, s = idx - h * NPAD;
            if (s < SDIM) {
                float a = 0.f;
#pragma unroll 8
                for (int j = 0; j < DHEAD; ++j) {
                    const int d = h * DHEAD + j;
                    a += qv[d] * (kt[d * RSTRIDE + s] + bkl[d]);
                }
                a *= SCALE;
                if (s == CENTER) a += MASK_VAL;
                sc[h * NPAD + s] = a;
            }
        }
        __syncthreads();

        // ---- softmax ----
        if (tid < HEADS) {
            float m = -3.4e38f;
            for (int s = 0; s < SDIM; ++s) m = fmaxf(m, sc[tid * NPAD + s]);
            float sum = 0.f;
            for (int s = 0; s < SDIM; ++s) sum += __expf(sc[tid * NPAD + s] - m);
            st[tid * 2]     = m;
            st[tid * 2 + 1] = 1.f / sum;
        }
        __syncthreads();
        for (int idx = tid; idx < HEADS * NPAD; idx += 256) {
            const int h = idx / NPAD, s = idx - h * NPAD;
            if (s < SDIM)
                sc[h * NPAD + s] = __expf(sc[h * NPAD + s] - st[h * 2]) * st[h * 2 + 1];
        }
        __syncthreads();

        // ---- surround, central, gate ----
        float central = 0.f, surround = 0.f;
        if (tid < DKDIM) {
            const int h = tid >> 5;
            float a = 0.f;
#pragma unroll 3
            for (int s = 0; s < SDIM; ++s)
                a += sc[h * NPAD + s] * (vt[tid * RSTRIDE + s] + bvl[tid]);
            surround = a;
            central  = vt[tid * RSTRIDE + CENTER] + bvl[tid];
            red[tid] = Wg[tid] * (central - surround);
        }
        __syncthreads();
        if (tid == 0) {
            float t = bg[0];
            for (int i = 0; i < DKDIM; ++i) t += red[i];
            ((float*)(smem + OFF_GATE))[0] = 1.f / (1.f + __expf(-t));
        }
        __syncthreads();
        if (tid < DKDIM) {
            const float g = ((float*)(smem + OFF_GATE))[0];
            out[(size_t)b * DKDIM + tid] = central + g * surround;
        }
        __syncthreads();
    }
}

extern "C" void kernel_launch(void* const* d_in, const int* in_sizes, int n_in,
                              void* d_out, int out_size, void* d_ws, size_t ws_size,
                              hipStream_t stream) {
    const float* x  = (const float*)d_in[0];
    const float* Wk = (const float*)d_in[1];
    const float* bk = (const float*)d_in[2];
    const float* Wv = (const float*)d_in[3];
    const float* bv = (const float*)d_in[4];
    const float* Wq = (const float*)d_in[5];
    const float* bq = (const float*)d_in[6];
    const float* Wg = (const float*)d_in[7];
    const float* bg = (const float*)d_in[8];
    float* out = (float*)d_out;

    const int batches = in_sizes[0] / XELEMS;           // 8192
    const int grid = batches < NBLOCKS ? batches : NBLOCKS;
    surround_attn_kernel<<<dim3(grid), dim3(256), 0, stream>>>(
        x, Wk, bk, Wv, bv, Wq, bq, Wg, bg, out, batches);
}
